// MambaBlock_19129784336931
// MI455X (gfx1250) — compile-verified, matrix operand order checked by reading, not run on hardware
//
#include <hip/hip_runtime.h>

// ---------------------------------------------------------------------------
// Mamba block forward for MI455X (gfx1250, wave32, WMMA).
// D_MODEL=1024, D_STATE=16, D_CONV=4, D_INNER=2048, BATCH=2, SEQ=2048.
// Big GEMMs use v_wmma_f32_16x16x32_bf16 (bf16 in, f32 accumulate).
// ---------------------------------------------------------------------------

#define D_MODEL 1024
#define D_STATE 16
#define D_CONV  4
#define D_INNER 2048
#define BATCH   2
#define SEQ     2048
#define NROWS   (BATCH * SEQ)          // 4096 (b,s) rows
#define E2      (2 * D_INNER)          // 4096
#define NSSM    (2 * D_STATE + 1)      // 33

typedef __attribute__((ext_vector_type(16))) __bf16 v16bf;
typedef __attribute__((ext_vector_type(8)))  float  v8f;

// float -> bf16 (round-to-nearest-even), stored as raw u16
__device__ __forceinline__ unsigned short f2bf(float f) {
    unsigned int u = __float_as_uint(f);
    u += 0x7FFFu + ((u >> 16) & 1u);
    return (unsigned short)(u >> 16);
}

// ---------------------------------------------------------------------------
// Elementwise f32 -> bf16 conversion (for weights)
// ---------------------------------------------------------------------------
__global__ void k_f32_to_bf16(const float* __restrict__ src,
                              unsigned short* __restrict__ dst, int n) {
    int i = blockIdx.x * blockDim.x + threadIdx.x;
    if (i < n) dst[i] = f2bf(src[i]);
}

// ---------------------------------------------------------------------------
// LayerNorm: one block per (b,s) row of 1024; writes bf16 activations
// ---------------------------------------------------------------------------
__global__ void k_layernorm(const float* __restrict__ x,
                            const float* __restrict__ w,
                            const float* __restrict__ b,
                            unsigned short* __restrict__ xn_bf) {
    __shared__ float s_sum[256], s_sq[256];
    const int row = blockIdx.x;
    const int tid = threadIdx.x;
    const float* xr = x + (size_t)row * D_MODEL;

    float sum = 0.f, sq = 0.f;
    #pragma unroll
    for (int c = tid; c < D_MODEL; c += 256) {
        float v = xr[c];
        sum += v; sq += v * v;
    }
    s_sum[tid] = sum; s_sq[tid] = sq;
    __syncthreads();
    for (int s = 128; s > 0; s >>= 1) {
        if (tid < s) { s_sum[tid] += s_sum[tid + s]; s_sq[tid] += s_sq[tid + s]; }
        __syncthreads();
    }
    const float mu  = s_sum[0] * (1.0f / D_MODEL);
    const float var = s_sq[0]  * (1.0f / D_MODEL) - mu * mu;
    const float rs  = rsqrtf(var + 1e-5f);

    #pragma unroll
    for (int c = tid; c < D_MODEL; c += 256) {
        float v = (xr[c] - mu) * rs * w[c] + b[c];
        xn_bf[(size_t)row * D_MODEL + c] = f2bf(v);
    }
}

// ---------------------------------------------------------------------------
// bf16 WMMA GEMM:  C[M,N] = A[M,K] * B[N,K]^T  (both K-contiguous, bf16)
// Workgroup = 256 threads = 8 waves; WG tile 128x128; wave tile 32x64.
// Fragment loading exploits the 16x16x32 bf16 VGPR layout:
//   lane p<16  -> index p,    K halves {k..k+7, k+16..k+23}
//   lane p>=16 -> index p-16, K halves {k+8..k+15, k+24..k+31}
// i.e. two b128 loads per lane per fragment, straight from global.
// Optional residual add in epilogue (resid==nullptr to skip).
// ---------------------------------------------------------------------------
__device__ __forceinline__ v16bf load_frag(const unsigned short* __restrict__ base,
                                           int idx0, int ld, int kbase, int lane) {
    const int r  = idx0 + (lane & 15);
    const int k0 = kbase + ((lane >> 4) << 3);     // +8 halves for lanes 16..31
    const unsigned short* p = base + (size_t)r * ld + k0;
    union { uint4 u[2]; v16bf v; } u;
    u.u[0] = *(const uint4*)(p);
    u.u[1] = *(const uint4*)(p + 16);
    return u.v;
}

__device__ __forceinline__ v8f wmma_bf16(v16bf a, v16bf b, v8f c) {
    // (neg_a, A, neg_b, B, c_mod, C, reuse_a, reuse_b)
    return __builtin_amdgcn_wmma_f32_16x16x32_bf16(false, a, false, b,
                                                   (short)0, c, false, false);
}

__global__ __launch_bounds__(256)
void k_gemm_bf16(const unsigned short* __restrict__ A,   // [M,K] bf16
                 const unsigned short* __restrict__ B,   // [N,K] bf16
                 float* __restrict__ C,                   // [M,N] f32
                 const float* __restrict__ resid,         // [M,N] or nullptr
                 int M, int N, int K) {
    const int lane = threadIdx.x & 31;
    const int w    = threadIdx.x >> 5;                    // 0..7
    const int m0   = blockIdx.y * 128 + (w & 3) * 32;     // wave row base
    const int n0   = blockIdx.x * 128 + (w >> 2) * 64;    // wave col base

    v8f acc[2][4];
    #pragma unroll
    for (int i = 0; i < 2; ++i)
        #pragma unroll
        for (int j = 0; j < 4; ++j) acc[i][j] = {};

    for (int k = 0; k < K; k += 32) {
        v16bf a0 = load_frag(A, m0,      K, k, lane);
        v16bf a1 = load_frag(A, m0 + 16, K, k, lane);
        v16bf b0 = load_frag(B, n0,      K, k, lane);
        v16bf b1 = load_frag(B, n0 + 16, K, k, lane);
        v16bf b2 = load_frag(B, n0 + 32, K, k, lane);
        v16bf b3 = load_frag(B, n0 + 48, K, k, lane);
        acc[0][0] = wmma_bf16(a0, b0, acc[0][0]);
        acc[0][1] = wmma_bf16(a0, b1, acc[0][1]);
        acc[0][2] = wmma_bf16(a0, b2, acc[0][2]);
        acc[0][3] = wmma_bf16(a0, b3, acc[0][3]);
        acc[1][0] = wmma_bf16(a1, b0, acc[1][0]);
        acc[1][1] = wmma_bf16(a1, b1, acc[1][1]);
        acc[1][2] = wmma_bf16(a1, b2, acc[1][2]);
        acc[1][3] = wmma_bf16(a1, b3, acc[1][3]);
    }

    // Epilogue: C/D layout -> VGPR r: lanes 0..15 = row r, lanes 16..31 = row r+8
    const int colL = lane & 15;
    const int rAdd = (lane >> 4) * 8;
    #pragma unroll
    for (int i = 0; i < 2; ++i) {
        #pragma unroll
        for (int j = 0; j < 4; ++j) {
            #pragma unroll
            for (int r = 0; r < 8; ++r) {
                int row = m0 + i * 16 + r + rAdd;
                int col = n0 + j * 16 + colL;
                size_t o = (size_t)row * N + col;
                float v = acc[i][j][r];
                if (resid) v += resid[o];
                C[o] = v;
            }
        }
    }
}

// ---------------------------------------------------------------------------
// Depthwise causal conv (D_CONV=4) + bias + SiLU.
// x_b lives in xz[..., 0:2048]; output xc is dense [row, 2048].
// ---------------------------------------------------------------------------
__global__ void k_conv_silu(const float* __restrict__ xz,
                            const float* __restrict__ cw,   // [D_INNER,4]
                            const float* __restrict__ cb,   // [D_INNER]
                            float* __restrict__ xc) {
    int i = blockIdx.x * blockDim.x + threadIdx.x;
    if (i >= NROWS * D_INNER) return;
    const int d = i & (D_INNER - 1);
    const int s = (i / D_INNER) & (SEQ - 1);
    const int b = i / (D_INNER * SEQ);

    float acc = cb[d];
    #pragma unroll
    for (int j = 0; j < D_CONV; ++j) {
        int sj = s - (D_CONV - 1) + j;
        if (sj >= 0)
            acc += xz[((size_t)(b * SEQ + sj)) * E2 + d] * cw[d * D_CONV + j];
    }
    float sig = 1.0f / (1.0f + __expf(-acc));
    xc[(size_t)i] = acc * sig;
}

// ---------------------------------------------------------------------------
// ssm projection: ssm[row, 0:33] = xc[row, :] . W_x[e, :]
// One block per row; row staged in LDS; 33 tree-reductions.
// ---------------------------------------------------------------------------
__global__ void k_ssm_proj(const float* __restrict__ xc,
                           const float* __restrict__ Wx,    // [33, D_INNER]
                           float* __restrict__ ssm) {
    __shared__ float srow[D_INNER];
    __shared__ float red[256];
    const int row = blockIdx.x;
    const int tid = threadIdx.x;
    for (int t = tid; t < D_INNER; t += 256)
        srow[t] = xc[(size_t)row * D_INNER + t];
    __syncthreads();
    for (int e = 0; e < NSSM; ++e) {
        const float* wr = Wx + (size_t)e * D_INNER;
        float p = 0.f;
        for (int t = tid; t < D_INNER; t += 256) p += srow[t] * wr[t];
        red[tid] = p;
        __syncthreads();
        for (int s = 128; s > 0; s >>= 1) {
            if (tid < s) red[tid] += red[tid + s];
            __syncthreads();
        }
        if (tid == 0) ssm[(size_t)row * NSSM + e] = red[0];
        __syncthreads();
    }
}

// ---------------------------------------------------------------------------
// Selective scan: 1 thread per (b,d) channel, h[16] in registers, serial over
// SEQ. Per-step B/C/delta_raw (33 floats, shared by all channels of a batch)
// staged in LDS. Fuses softplus(delta), D skip, SiLU(z) gate; emits y in bf16.
// ---------------------------------------------------------------------------
__global__ __launch_bounds__(256)
void k_scan(const float* __restrict__ ssm,     // [NROWS, 33]
            const float* __restrict__ xc,      // [NROWS, D_INNER]
            const float* __restrict__ xz,      // z = xz[..., 2048:4096]
            const float* __restrict__ Wdt,     // [D_INNER]
            const float* __restrict__ bdt,     // [D_INNER]
            const float* __restrict__ Alog,    // [D_INNER, D_STATE]
            const float* __restrict__ Dvec,    // [D_INNER]
            unsigned short* __restrict__ y_bf) // [NROWS, D_INNER]
{
    __shared__ float sBC[NSSM];
    const int tid = threadIdx.x;
    const int b   = blockIdx.x >> 3;                    // 8 blocks per batch
    const int d   = (blockIdx.x & 7) * 256 + tid;

    float A[D_STATE], h[D_STATE];
    #pragma unroll
    for (int n = 0; n < D_STATE; ++n) {
        A[n] = -__expf(Alog[(size_t)d * D_STATE + n]);
        h[n] = 0.f;
    }
    const float wdt = Wdt[d], bd = bdt[d], Dd = Dvec[d];

    for (int s = 0; s < SEQ; ++s) {
        const size_t row = (size_t)b * SEQ + s;
        if (tid < NSSM) sBC[tid] = ssm[row * NSSM + tid];
        __syncthreads();

        const float draw = sBC[0] * wdt + bd;
        const float delta = (draw > 20.f) ? draw : log1pf(__expf(draw));
        const float xt = xc[row * D_INNER + d];
        const float dx = delta * xt;

        float y = 0.f;
        #pragma unroll
        for (int n = 0; n < D_STATE; ++n) {
            float dA = __expf(delta * A[n]);
            h[n] = dA * h[n] + dx * sBC[1 + n];
            y += h[n] * sBC[1 + D_STATE + n];
        }
        y += xt * Dd;
        const float z = xz[row * E2 + D_INNER + d];
        y *= z / (1.0f + __expf(-z));
        y_bf[row * D_INNER + d] = f2bf(y);
        __syncthreads();
    }
}

// ---------------------------------------------------------------------------
// Host-side launcher
// ---------------------------------------------------------------------------
extern "C" void kernel_launch(void* const* d_in, const int* in_sizes, int n_in,
                              void* d_out, int out_size, void* d_ws, size_t ws_size,
                              hipStream_t stream) {
    (void)in_sizes; (void)n_in; (void)out_size; (void)ws_size;
    const float* x      = (const float*)d_in[0];
    const float* ln_w   = (const float*)d_in[1];
    const float* ln_b   = (const float*)d_in[2];
    const float* W_in   = (const float*)d_in[3];   // [4096, 1024]
    const float* conv_w = (const float*)d_in[4];   // [2048, 1, 4]
    const float* conv_b = (const float*)d_in[5];
    const float* W_x    = (const float*)d_in[6];   // [33, 2048]
    const float* W_dt   = (const float*)d_in[7];   // [2048, 1]
    const float* b_dt   = (const float*)d_in[8];
    const float* A_log  = (const float*)d_in[9];   // [2048, 16]
    const float* Dv     = (const float*)d_in[10];
    const float* W_out  = (const float*)d_in[11];  // [1024, 2048]
    float* out = (float*)d_out;

    // Workspace layout (bytes)
    char* ws = (char*)d_ws;
    unsigned short* xn_bf   = (unsigned short*)(ws);                           //  8 MiB
    unsigned short* win_bf  = (unsigned short*)(ws + (((size_t)8)   << 20));   //  8 MiB
    float*          xz      = (float*)         (ws + (((size_t)16)  << 20));   // 64 MiB
    float*          xc      = (float*)         (ws + (((size_t)80)  << 20));   // 32 MiB
    float*          ssm     = (float*)         (ws + (((size_t)112) << 20));   // ~0.6 MiB
    unsigned short* y_bf    = (unsigned short*)(ws + (((size_t)113) << 20));   // 16 MiB
    unsigned short* wout_bf = (unsigned short*)(ws + (((size_t)129) << 20));   //  4 MiB

    // 1) weight conversions to bf16
    {
        int n1 = E2 * D_MODEL;      // 4,194,304
        int n2 = D_MODEL * D_INNER; // 2,097,152
        k_f32_to_bf16<<<(n1 + 255) / 256, 256, 0, stream>>>(W_in, win_bf, n1);
        k_f32_to_bf16<<<(n2 + 255) / 256, 256, 0, stream>>>(W_out, wout_bf, n2);
    }

    // 2) LayerNorm -> bf16 activations
    k_layernorm<<<NROWS, 256, 0, stream>>>(x, ln_w, ln_b, xn_bf);

    // 3) In-projection GEMM: xz[4096,4096] = xn[4096,1024] * W_in^T
    {
        dim3 grid(E2 / 128, NROWS / 128);
        k_gemm_bf16<<<grid, 256, 0, stream>>>(xn_bf, win_bf, xz, nullptr,
                                              NROWS, E2, D_MODEL);
    }

    // 4) depthwise conv + SiLU
    {
        int n = NROWS * D_INNER;
        k_conv_silu<<<n / 256, 256, 0, stream>>>(xz, conv_w, conv_b, xc);
    }

    // 5) ssm projection (N=33, VALU reduction)
    k_ssm_proj<<<NROWS, 256, 0, stream>>>(xc, W_x, ssm);

    // 6) selective scan + gate, emits y in bf16
    k_scan<<<(BATCH * D_INNER) / 256, 256, 0, stream>>>(
        ssm, xc, xz, W_dt, b_dt, A_log, Dv, y_bf);

    // 7) Out-projection GEMM + residual: out[4096,1024] = y[4096,2048]*W_out^T + x
    {
        dim3 grid(D_MODEL / 128, NROWS / 128);
        k_gemm_bf16<<<grid, 256, 0, stream>>>(y_bf, wout_bf, out, x,
                                              NROWS, D_MODEL, D_INNER);
    }
}
